// VRWKV_SpatialMix_11098195493456
// MI455X (gfx1250) — matrix operand order repaired and checked by reading, not compiled, not run on hardware
//
#include <hip/hip_runtime.h>
#include <hip/hip_bf16.h>
#include <cmath>

typedef __attribute__((ext_vector_type(16))) _Float16 v16h;
typedef __attribute__((ext_vector_type(8)))  _Float16 v8h;
typedef __attribute__((ext_vector_type(8)))  float    v8f;

#define CH 192
#define HH 96
#define WW 96
#define TT (HH*WW)
#define MIN_LOG (-1e38f)

__device__ inline v16h cat16(v8h lo, v8h hi) {
    v16h r;
#pragma unroll
    for (int i = 0; i < 8; ++i) { r[i] = lo[i]; r[8 + i] = hi[i]; }
    return r;
}

// ---------------------------------------------------------------------------
// Kernel 0: fuse alpha/w1/w3/w5 into one depthwise 5x5 weight per channel.
// ---------------------------------------------------------------------------
__global__ void combine_weights_kernel(const float* __restrict__ alpha,
                                       const float* __restrict__ w1,
                                       const float* __restrict__ w3,
                                       const float* __restrict__ w5,
                                       float* __restrict__ comb) {
    int i = blockIdx.x * blockDim.x + threadIdx.x;
    if (i >= CH * 25) return;
    int c = i / 25, r = i % 25, p = r / 5, q = r % 5;
    float v = alpha[3] * w5[i];
    if (p >= 1 && p <= 3 && q >= 1 && q <= 3)
        v += alpha[2] * w3[c * 9 + (p - 1) * 3 + (q - 1)];
    if (p == 2 && q == 2)
        v += alpha[0] + alpha[1] * w1[c];
    comb[i] = v;
}

// ---------------------------------------------------------------------------
// Kernel 0b: one-time f32 -> f16 conversion of a weight matrix.
// ---------------------------------------------------------------------------
__global__ void f32_to_f16_kernel(const float* __restrict__ src,
                                  _Float16* __restrict__ dst, int n) {
    int i = blockIdx.x * blockDim.x + threadIdx.x;
    if (i < n) dst[i] = (_Float16)src[i];
}

// ---------------------------------------------------------------------------
// Kernel 1: depthwise 5x5 conv, channel-last; accumulate f32, store f16
// (the conv output feeds only the f16 WMMA projections).
// ---------------------------------------------------------------------------
__global__ void omni_shift_kernel(const float* __restrict__ x,
                                  const float* __restrict__ comb,
                                  _Float16* __restrict__ xsH, int Bn) {
    size_t tid = (size_t)blockIdx.x * blockDim.x + threadIdx.x;
    size_t total = (size_t)Bn * TT * CH;
    if (tid >= total) return;
    int c = (int)(tid % CH);
    size_t s = tid / CH;
    int w = (int)(s % WW);
    int h = (int)((s / WW) % HH);
    int b = (int)(s / ((size_t)WW * HH));
    float acc = 0.f;
#pragma unroll
    for (int dy = 0; dy < 5; ++dy) {
        int hy = h + dy - 2;
        if (hy < 0 || hy >= HH) continue;
#pragma unroll
        for (int dx = 0; dx < 5; ++dx) {
            int wx = w + dx - 2;
            if (wx < 0 || wx >= WW) continue;
            acc += comb[c * 25 + dy * 5 + dx] *
                   x[((size_t)b * TT + (size_t)hy * WW + wx) * CH + c];
        }
    }
    xsH[tid] = (_Float16)acc;
}

// ---------------------------------------------------------------------------
// Kernel 2: fused triple projection with LDS-staged f16 weights.
// Block = 8 waves = 8 m-tiles at one n-tile; per-block B reuse is 8x.
// A fragment (16x32 f16): lane l<16 row l, K=0..15;
//                         lane l+16      K=16..31.
// B fragment (32x16 f16): lane n<16 col n, K=0..15; lane n+16, K=16..31.
// ---------------------------------------------------------------------------
__global__ void proj_gemm_kernel(const _Float16* __restrict__ A,
                                 const _Float16* __restrict__ WkH,
                                 const _Float16* __restrict__ WvH,
                                 const _Float16* __restrict__ WrH,
                                 float* __restrict__ Ko,
                                 float* __restrict__ Vo,
                                 float* __restrict__ Ro, int M) {
    __shared__ __align__(16) _Float16 blds[3][16][CH];   // 18 KB
    int nBase = blockIdx.y * 16;
    const int K8 = CH / 8;                                // 24 chunks per row
    for (int idx = threadIdx.x; idx < 3 * 16 * K8; idx += blockDim.x) {
        int m  = idx / (16 * K8);
        int r  = idx - m * (16 * K8);
        int n  = r / K8;
        int k8 = (r - n * K8) * 8;
        const _Float16* s = (m == 0) ? WkH : (m == 1) ? WvH : WrH;
        *(v8h*)&blds[m][n][k8] = *(const v8h*)(s + (size_t)(nBase + n) * CH + k8);
    }
    __syncthreads();

    int lane = threadIdx.x & 31;
    int wave = threadIdx.x >> 5;
    int l  = lane & 15;
    int hs = lane >> 4;
    int mBase = (blockIdx.x * 8 + wave) * 16;             // M divisible by 128

    const _Float16* a_row = A + (size_t)(mBase + l) * CH;

    v8f ck = {}; v8f cv = {}; v8f cr = {};
#pragma unroll
    for (int k0 = 0; k0 < CH; k0 += 32) {
        v16h a = cat16(*(const v8h*)(a_row + k0 + hs * 16),
                       *(const v8h*)(a_row + k0 + hs * 16 + 8));
        v16h bK = cat16(*(const v8h*)&blds[0][l][k0 + hs * 16],
                        *(const v8h*)&blds[0][l][k0 + hs * 16 + 8]);
        v16h bV = cat16(*(const v8h*)&blds[1][l][k0 + hs * 16],
                        *(const v8h*)&blds[1][l][k0 + hs * 16 + 8]);
        v16h bR = cat16(*(const v8h*)&blds[2][l][k0 + hs * 16],
                        *(const v8h*)&blds[2][l][k0 + hs * 16 + 8]);
        ck = __builtin_amdgcn_wmma_f32_16x16x32_f16(false, a, false, bK, (short)0, ck, false, false);
        cv = __builtin_amdgcn_wmma_f32_16x16x32_f16(false, a, false, bV, (short)0, cv, false, false);
        cr = __builtin_amdgcn_wmma_f32_16x16x32_f16(false, a, false, bR, (short)0, cr, false, false);
    }
    int col = nBase + l;
#pragma unroll
    for (int i = 0; i < 8; ++i) {
        size_t off = (size_t)(mBase + hs * 8 + i) * CH + col;
        Ko[off] = ck[i];
        Vo[off] = cv[i];
        Ro[off] = 1.f / (1.f + __expf(-cr[i]));
    }
}

// ---------------------------------------------------------------------------
// Kernel 3: output GEMM  y = AH @ Wo^T, AH = (sr .* v) precomputed in f16.
// ---------------------------------------------------------------------------
__global__ void out_gemm_kernel(const _Float16* __restrict__ AH,
                                const _Float16* __restrict__ WoH,
                                float* __restrict__ Out, int M) {
    __shared__ __align__(16) _Float16 wlds[16][CH];       // 6 KB
    int nBase = blockIdx.y * 16;
    const int K8 = CH / 8;
    for (int idx = threadIdx.x; idx < 16 * K8; idx += blockDim.x) {
        int n  = idx / K8;
        int k8 = (idx - n * K8) * 8;
        *(v8h*)&wlds[n][k8] = *(const v8h*)(WoH + (size_t)(nBase + n) * CH + k8);
    }
    __syncthreads();

    int lane = threadIdx.x & 31;
    int wave = threadIdx.x >> 5;
    int l  = lane & 15;
    int hs = lane >> 4;
    int mBase = (blockIdx.x * 8 + wave) * 16;

    const _Float16* a_row = AH + (size_t)(mBase + l) * CH;

    v8f acc = {};
#pragma unroll
    for (int k0 = 0; k0 < CH; k0 += 32) {
        v16h a = cat16(*(const v8h*)(a_row + k0 + hs * 16),
                       *(const v8h*)(a_row + k0 + hs * 16 + 8));
        v16h b = cat16(*(const v8h*)&wlds[l][k0 + hs * 16],
                       *(const v8h*)&wlds[l][k0 + hs * 16 + 8]);
        acc = __builtin_amdgcn_wmma_f32_16x16x32_f16(false, a, false, b, (short)0, acc, false, false);
    }
    int col = nBase + l;
#pragma unroll
    for (int i = 0; i < 8; ++i)
        Out[(size_t)(mBase + hs * 8 + i) * CH + col] = acc[i];
}

// ---------------------------------------------------------------------------
// WKV scan, chunked 3-phase parallel prefix (per-channel decay e^w constant,
// so chunk composition is:  state_out = e^{w*L} * state_in  (+)  chunk_sum).
// State (p, q, o) represents (p*e^o, q*e^o).
// ---------------------------------------------------------------------------
__global__ void wkv_phase1(const float* __restrict__ K,
                           const float* __restrict__ V,
                           const float* __restrict__ decay,
                           float* __restrict__ ps, float* __restrict__ qs,
                           float* __restrict__ os,
                           int Bn, int nCh, int L, int chunkStride, int elemStride) {
    int tid = blockIdx.x * blockDim.x + threadIdx.x;
    if (tid >= nCh * Bn * CH) return;
    int c = tid % CH;
    int rest = tid / CH;
    int b = rest % Bn;
    int cc = rest / Bn;
    float w = decay[c] / (float)TT;
    float p = 0.f, q = 0.f, o = MIN_LOG;
    for (int i = 0; i < L; ++i) {
        size_t pos = ((size_t)b * TT + (size_t)cc * chunkStride + (size_t)i * elemStride) * CH + c;
        float kt = K[pos], vt = V[pos];
        float no = fmaxf(w + o, kt);
        float A2 = __expf(w + o - no), B2 = __expf(kt - no);
        p = A2 * p + B2 * vt;
        q = A2 * q + B2;
        o = no;
    }
    size_t idx = ((size_t)cc * Bn + b) * CH + c;
    ps[idx] = p; qs[idx] = q; os[idx] = o;
}

__global__ void wkv_phase2(const float* __restrict__ decay,
                           float* __restrict__ ps, float* __restrict__ qs,
                           float* __restrict__ os,
                           int Bn, int nCh, int L) {
    int tid = blockIdx.x * blockDim.x + threadIdx.x;
    if (tid >= Bn * CH) return;
    int c = tid % CH;
    int b = tid / CH;
    float wL = (decay[c] / (float)TT) * (float)L;
    float cp = 0.f, cq = 0.f, co = MIN_LOG;
    for (int cc = 0; cc < nCh; ++cc) {
        size_t idx = ((size_t)cc * Bn + b) * CH + c;
        float sp = ps[idx], sq = qs[idx], so = os[idx];
        ps[idx] = cp; qs[idx] = cq; os[idx] = co;      // exclusive prefix
        float o1 = co + wL;
        float no = fmaxf(o1, so);
        float A2 = __expf(o1 - no), B2 = __expf(so - no);
        cp = A2 * cp + B2 * sp;
        cq = A2 * cq + B2 * sq;
        co = no;
    }
}

// Phase 3: replay chunk from its exclusive-prefix state and emit y.
// If rmul != nullptr (final scan): write f16 (sr*y) into aH instead of V,
// feeding the output GEMM directly (saves a full f32 round trip).
__global__ void wkv_phase3(const float* __restrict__ K,
                           float* __restrict__ V,
                           const float* __restrict__ decay,
                           const float* __restrict__ first,
                           const float* __restrict__ ps, const float* __restrict__ qs,
                           const float* __restrict__ os,
                           const float* __restrict__ rmul,
                           _Float16* __restrict__ aH,
                           int Bn, int nCh, int L, int chunkStride, int elemStride) {
    int tid = blockIdx.x * blockDim.x + threadIdx.x;
    if (tid >= nCh * Bn * CH) return;
    int c = tid % CH;
    int rest = tid / CH;
    int b = rest % Bn;
    int cc = rest / Bn;
    float w = decay[c] / (float)TT;
    float u = first[c] / (float)TT;
    size_t idx = ((size_t)cc * Bn + b) * CH + c;
    float p = ps[idx], q = qs[idx], o = os[idx];
    for (int i = 0; i < L; ++i) {
        size_t pos = ((size_t)b * TT + (size_t)cc * chunkStride + (size_t)i * elemStride) * CH + c;
        float kt = K[pos], vt = V[pos];
        float no = fmaxf(o, u + kt);
        float A = __expf(o - no), Bc = __expf(u + kt - no);
        float y = (A * p + Bc * vt) / (A * q + Bc);
        if (rmul) aH[pos] = (_Float16)(y * rmul[pos]);
        else      V[pos] = y;
        float no2 = fmaxf(w + o, kt);
        float A2 = __expf(w + o - no2), B2 = __expf(kt - no2);
        p = A2 * p + B2 * vt;
        q = A2 * q + B2;
        o = no2;
    }
}

// ---------------------------------------------------------------------------
extern "C" void kernel_launch(void* const* d_in, const int* in_sizes, int n_in,
                              void* d_out, int out_size, void* d_ws, size_t ws_size,
                              hipStream_t stream) {
    const float* x      = (const float*)d_in[0];
    const float* alpha  = (const float*)d_in[1];
    const float* w1     = (const float*)d_in[2];
    const float* w3     = (const float*)d_in[3];
    const float* w5     = (const float*)d_in[4];
    const float* Wk     = (const float*)d_in[5];
    const float* Wv     = (const float*)d_in[6];
    const float* Wr     = (const float*)d_in[7];
    const float* Wo     = (const float*)d_in[8];
    const float* sdecay = (const float*)d_in[9];
    const float* sfirst = (const float*)d_in[10];

    int Bn = in_sizes[0] / (TT * CH);          // 8
    size_t NE = (size_t)Bn * TT * CH;
    size_t SC = (size_t)HH * Bn * CH;          // chunk-state size (HH==WW)

    float* ws   = (float*)d_ws;
    float* comb = ws;                          // CH*25 (padded to 8192)
    float* kk   = ws + 8192;
    float* vv   = kk + NE;
    float* rr   = vv + NE;
    float* ps   = rr + NE;
    float* qs   = ps + SC;
    float* os   = qs + SC;
    _Float16* xsH = (_Float16*)(os + SC);
    _Float16* aH  = xsH + NE;
    _Float16* wkH = aH + NE;
    _Float16* wvH = wkH + (size_t)CH * CH;
    _Float16* wrH = wvH + (size_t)CH * CH;
    _Float16* woH = wrH + (size_t)CH * CH;

    combine_weights_kernel<<<(CH * 25 + 255) / 256, 256, 0, stream>>>(alpha, w1, w3, w5, comb);
    int nw = CH * CH;
    f32_to_f16_kernel<<<(nw + 255) / 256, 256, 0, stream>>>(Wk, wkH, nw);
    f32_to_f16_kernel<<<(nw + 255) / 256, 256, 0, stream>>>(Wv, wvH, nw);
    f32_to_f16_kernel<<<(nw + 255) / 256, 256, 0, stream>>>(Wr, wrH, nw);
    f32_to_f16_kernel<<<(nw + 255) / 256, 256, 0, stream>>>(Wo, woH, nw);

    omni_shift_kernel<<<(int)((NE + 255) / 256), 256, 0, stream>>>(x, comb, xsH, Bn);

    int M = Bn * TT;                            // 73728, divisible by 128
    dim3 gg(M / 128, CH / 16);
    proj_gemm_kernel<<<gg, 256, 0, stream>>>(xsH, wkH, wvH, wrH, kk, vv, rr, M);

    // scan 0: row-major. chunks = HH rows of length WW; idx = cc*WW + i
    {
        int nCh = HH, L = WW;
        int n1 = nCh * Bn * CH;
        wkv_phase1<<<(n1 + 255) / 256, 256, 0, stream>>>(kk, vv, sdecay, ps, qs, os, Bn, nCh, L, WW, 1);
        wkv_phase2<<<(Bn * CH + 255) / 256, 256, 0, stream>>>(sdecay, ps, qs, os, Bn, nCh, L);
        wkv_phase3<<<(n1 + 255) / 256, 256, 0, stream>>>(kk, vv, sdecay, sfirst, ps, qs, os,
                                                         nullptr, nullptr, Bn, nCh, L, WW, 1);
    }
    // scan 1: column-major. chunks = WW columns of length HH; idx = i*WW + cc.
    // Phase 3 writes the f16 (sr*y) operand for the output GEMM directly.
    {
        int nCh = WW, L = HH;
        int n1 = nCh * Bn * CH;
        wkv_phase1<<<(n1 + 255) / 256, 256, 0, stream>>>(kk, vv, sdecay + CH, ps, qs, os, Bn, nCh, L, 1, WW);
        wkv_phase2<<<(Bn * CH + 255) / 256, 256, 0, stream>>>(sdecay + CH, ps, qs, os, Bn, nCh, L);
        wkv_phase3<<<(n1 + 255) / 256, 256, 0, stream>>>(kk, vv, sdecay + CH, sfirst + CH, ps, qs, os,
                                                         rr, aH, Bn, nCh, L, 1, WW);
    }

    out_gemm_kernel<<<gg, 256, 0, stream>>>(aH, woH, (float*)d_out, M);
}
